// DINEncoder_23940147708093
// MI455X (gfx1250) — compile-verified, matrix operand order checked by reading, not run on hardware
//
#include <hip/hip_runtime.h>
#include <hip/hip_bf16.h>

// ---------------------------------------------------------------------------
// DIN local-activation attention, fused, for MI455X (gfx1250, wave32, WMMA).
// B=8, N=512, D=64, H=64.
//   P0b: hk[k,h]   = x @ (Wk-Wd)^T           (WMMA, once per workgroup)
//   P1 : pre-act   = (x_q * Wm) @ x_k^T + hk + hq + b1, PReLU, .W2 -> scores
//        (8x v_wmma_f32_16x16x32_bf16 per k-tile, causal k-tiles only)
//   P2 : encoded   = scores @ x              (WMMA, causally truncated K)
// Heavy q-tiles launch first (reversed mapping) to shorten the causal tail.
// ---------------------------------------------------------------------------

typedef __attribute__((ext_vector_type(16))) __bf16 v16bf;
typedef __attribute__((ext_vector_type(8)))  __bf16 v8bf;
typedef __attribute__((ext_vector_type(8)))  float  v8f;

#define B_   8
#define N_   512
#define D_   64
#define H_   64
#define XS   72    // LDS row stride (bf16) for [*,64] tiles: 144B -> conflict-free
#define TS   520   // LDS row stride (bf16) for [*,512] tiles: 1040B -> conflict-free

union FragU { v16bf v; v8bf h[2]; };

// One 16x32 (A) / 32x16 (B) bf16 WMMA fragment from LDS.
// ISA layout: lane<16 holds K {0..7,16..23}, lane>=16 holds K {8..15,24..31}
// of row/col (lane & 15). rowptr already includes row*stride + K-chunk base.
__device__ inline v16bf load_frag(const __bf16* rowptr, int lane) {
  const int koff = (lane & 16) ? 8 : 0;
  FragU u;
  u.h[0] = *(const v8bf*)(rowptr + koff);        // ds_load_b128
  u.h[1] = *(const v8bf*)(rowptr + koff + 16);   // ds_load_b128
  return u.v;
}

// Convert 8 consecutive f32 (two float4) to a v8bf.
__device__ inline v8bf cvt8(const float4 f0, const float4 f1) {
  v8bf r;
  r[0] = (__bf16)f0.x; r[1] = (__bf16)f0.y; r[2] = (__bf16)f0.z; r[3] = (__bf16)f0.w;
  r[4] = (__bf16)f1.x; r[5] = (__bf16)f1.y; r[6] = (__bf16)f1.z; r[7] = (__bf16)f1.w;
  return r;
}

__global__ __launch_bounds__(256)
void din_encoder_kernel(const float* __restrict__ UE,   // [B,N,D]
                        const float* __restrict__ VM,   // [B,N]
                        const float* __restrict__ W1,   // [H,4D]
                        const float* __restrict__ b1,   // [H]
                        const float* __restrict__ pa,   // [1]
                        const float* __restrict__ W2,   // [1,H]
                        const float* __restrict__ b2,   // [1]
                        float* __restrict__ out)        // [B,N,D]
{
  __shared__ __align__(16) __bf16 sX  [N_ * XS];   // x[b] bf16           73728 B
  __shared__ __align__(16) __bf16 sXT [D_ * TS];   // x[b]^T bf16         66560 B
  __shared__ __align__(16) __bf16 sHK [N_ * XS];   // hk[k,h] bf16        73728 B
  __shared__ __align__(16) __bf16 sWm [H_ * XS];   // Wm block             9216 B
  __shared__ __align__(16) __bf16 sWkd[H_ * XS];   // (Wk-Wd) block        9216 B
  __shared__ __align__(16) __bf16 sSB [16 * TS];   // scores bf16         16640 B
  __shared__ float sHqb[16 * H_];                  // hq+b1                4096 B
  __shared__ float sW2[H_];

  const int tid  = threadIdx.x;
  const int lane = tid & 31;
  const int wave = tid >> 5;
  const int m16  = lane & 15;
  const int b    = blockIdx.x >> 5;          // batch
  const int iqt  = 31 - (blockIdx.x & 31);   // q-tile, heaviest first
  const int q0   = iqt << 4;

  const float prelu_a = pa[0];
  const float bias2   = b2[0];

  // ---------------- P0: stage operands (vectorized) ------------------------
  const float* Xg = UE + (size_t)b * (N_ * D_);
  for (int idx = tid; idx < (N_ * D_) / 8; idx += 256) {   // 16 iters/thread
    int k  = idx >> 3;
    int d8 = (idx & 7) << 3;
    const float4* g = (const float4*)(Xg + k * D_ + d8);
    v8bf v = cvt8(g[0], g[1]);
    *(v8bf*)(sX + k * XS + d8) = v;                        // ds_store_b128
    for (int e = 0; e < 8; ++e)                            // transposed copy
      sXT[(d8 + e) * TS + k] = v[e];
  }
  for (int idx = tid; idx < (H_ * D_) / 8; idx += 256) {   // 2 iters/thread
    int h  = idx >> 3;
    int d8 = (idx & 7) << 3;
    const float4* wm = (const float4*)(W1 + h * (4 * D_) + 3 * D_ + d8);
    const float4* wk = (const float4*)(W1 + h * (4 * D_) + D_     + d8);
    const float4* wd = (const float4*)(W1 + h * (4 * D_) + 2 * D_ + d8);
    *(v8bf*)(sWm + h * XS + d8) = cvt8(wm[0], wm[1]);
    float4 k0 = wk[0], k1 = wk[1], d0 = wd[0], d1 = wd[1];
    float4 s0 = make_float4(k0.x - d0.x, k0.y - d0.y, k0.z - d0.z, k0.w - d0.w);
    float4 s1 = make_float4(k1.x - d1.x, k1.y - d1.y, k1.z - d1.z, k1.w - d1.w);
    *(v8bf*)(sWkd + h * XS + d8) = cvt8(s0, s1);
  }
  if (tid < H_) sW2[tid] = W2[tid];
  {
    v8bf z;
    for (int e = 0; e < 8; ++e) z[e] = (__bf16)0.0f;
    for (int idx = tid; idx < (16 * TS) / 8; idx += 256)   // zero scores
      ((v8bf*)sSB)[idx] = z;
  }
  // hq[q,h] = b1[h] + x[q,:] . (Wq + Wd)[h,:]   (tiny f32 precompute)
  for (int idx = tid; idx < 16 * H_; idx += 256) {
    int q = idx >> 6, h = idx & 63;
    const float* xq  = Xg + (size_t)(q0 + q) * D_;
    const float* w1r = W1 + h * (4 * D_);
    float acc = b1[h];
    for (int d = 0; d < D_; ++d) acc += xq[d] * (w1r[d] + w1r[2 * D_ + d]);
    sHqb[idx] = acc;
  }
  __syncthreads();

  // ---------------- P0b: hk = x @ (Wk-Wd)^T via WMMA (q-independent) ------
  // Each wave: 4 k-tiles; per k-tile the 4 h-chunks are independent WMMA
  // chains sharing one pair of A-fragments (fills WMMA->VALU hazard slots).
  {
    // Hoist the 8 (Wk-Wd) B-fragments: reused for all 4 k-tiles.
    v16bf wf[4][2];
    for (int j = 0; j < 4; ++j)
      for (int cch = 0; cch < 2; ++cch)
        wf[j][cch] = load_frag(sWkd + (16 * j + m16) * XS + 32 * cch, lane);

    for (int u = 0; u < 4; ++u) {
      const int kt = (u << 3) + wave;                      // 32 k-tiles / 8 waves
      v16bf a0 = load_frag(sX + (16 * kt + m16) * XS,      lane);
      v16bf a1 = load_frag(sX + (16 * kt + m16) * XS + 32, lane);
      v8f c[4];
      for (int j = 0; j < 4; ++j) {
        v8f z = {0.f, 0.f, 0.f, 0.f, 0.f, 0.f, 0.f, 0.f};
        z = __builtin_amdgcn_wmma_f32_16x16x32_bf16(false, a0, false, wf[j][0], (short)0, z, false, false);
        z = __builtin_amdgcn_wmma_f32_16x16x32_bf16(false, a1, false, wf[j][1], (short)0, z, false, false);
        c[j] = z;
      }
      for (int j = 0; j < 4; ++j) {
        const int hcol = 16 * j + m16;
        for (int v = 0; v < 8; ++v) {
          int k = 16 * kt + v + ((lane >> 4) << 3);
          sHK[k * XS + hcol] = (__bf16)c[j][v];
        }
      }
    }
  }
  __syncthreads();

  // ---------------- P1: main WMMA loop ------------------------------------
  for (int q = wave; q < 16; q += 8) {        // 2 queries per wave, uniform
    const int qg = q0 + q;

    // A-frags: a[(q,h),d] = x[q,d] * Wm[h,d]  (x[q,:] is lane-uniform).
    v16bf af[4][2];
    for (int cch = 0; cch < 2; ++cch) {
      v16bf xq = load_frag(sX + qg * XS + 32 * cch, lane);
      for (int j = 0; j < 4; ++j) {
        v16bf wm = load_frag(sWm + (16 * j + m16) * XS + 32 * cch, lane);
        v16bf a;
        for (int e = 0; e < 16; ++e)
          a[e] = (__bf16)((float)wm[e] * (float)xq[e]);
        af[j][cch] = a;
      }
    }

    for (int kt = 0; kt <= iqt; ++kt) {       // causal k-tiles only
      const __bf16* brow = sX + (16 * kt + m16) * XS;
      v16bf bf0 = load_frag(brow,      lane);
      v16bf bf1 = load_frag(brow + 32, lane);

      v8f acc[4];
      for (int j = 0; j < 4; ++j) {
        v8f c = {0.f, 0.f, 0.f, 0.f, 0.f, 0.f, 0.f, 0.f};
        c = __builtin_amdgcn_wmma_f32_16x16x32_bf16(false, af[j][0], false, bf0, (short)0, c, false, false);
        c = __builtin_amdgcn_wmma_f32_16x16x32_bf16(false, af[j][1], false, bf1, (short)0, c, false, false);
        acc[j] = c;
      }

      // Epilogue: + hk[k,h] (LDS bf16) + hq+b1 (f32), PReLU, .W2, reduce h.
      // C-layout: (vgpr v, lane l) -> h = 16j + v + 8*(l>=16), k = k0 + (l&15).
      float s = 0.f;
      const int hbase = (lane >> 4) << 3;
      const __bf16* hkrow = sHK + (16 * kt + m16) * XS;  // this lane's k row
      for (int j = 0; j < 4; ++j) {
        v8bf hk8 = *(const v8bf*)(hkrow + 16 * j + hbase);  // ds_load_b128
        for (int v = 0; v < 8; ++v) {
          int h = 16 * j + hbase + v;
          float val = acc[j][v] + (float)hk8[v] + sHqb[q * H_ + h];
          val = (val > 0.f) ? val : prelu_a * val;           // PReLU
          s += val * sW2[h];
        }
      }
      s += __shfl_xor(s, 16, 32);             // fold lane-halves (h split)

      if (lane < 16) {
        int kk = 16 * kt + lane;
        float sc = s + bias2;
        sc = (kk <= qg) ? sc * VM[b * N_ + kk] : 0.f;  // causal + padding
        sSB[q * TS + kk] = (__bf16)sc;
      }
    }
  }
  __syncthreads();

  // ---------------- P2: encoded = scores @ x via WMMA ---------------------
  // Output [16q x 64d]: 4 d-tiles on waves 0..3; K truncated at causal bound.
  if (wave < 4) {
    const int d0  = wave << 4;
    const int kch = (16 * (iqt + 1) + 31) >> 5;    // # of 32-wide K chunks
    v8f c = {0.f, 0.f, 0.f, 0.f, 0.f, 0.f, 0.f, 0.f};
    for (int kc = 0; kc < kch; ++kc) {
      v16bf a = load_frag(sSB + m16 * TS + 32 * kc, lane);          // rows=q
      v16bf x = load_frag(sXT + (d0 + m16) * TS + 32 * kc, lane);   // cols=d
      c = __builtin_amdgcn_wmma_f32_16x16x32_bf16(false, a, false, x, (short)0, c, false, false);
    }
    float* ob = out + ((size_t)b * N_ + q0) * D_;
    for (int v = 0; v < 8; ++v) {
      int q = v + ((lane >> 4) << 3);
      ob[q * D_ + d0 + m16] = c[v];
    }
  }
}

extern "C" void kernel_launch(void* const* d_in, const int* in_sizes, int n_in,
                              void* d_out, int out_size, void* d_ws, size_t ws_size,
                              hipStream_t stream) {
  // inputs: 0 past_lengths (unused), 1 user_embeddings, 2 valid_mask,
  //         3 W1, 4 b1, 5 prelu_a, 6 W2, 7 b2
  const float* UE = (const float*)d_in[1];
  const float* VM = (const float*)d_in[2];
  const float* W1 = (const float*)d_in[3];
  const float* b1 = (const float*)d_in[4];
  const float* pa = (const float*)d_in[5];
  const float* W2 = (const float*)d_in[6];
  const float* b2 = (const float*)d_in[7];
  float* out = (float*)d_out;

  dim3 grid(B_ * (N_ / 16));   // 256 workgroups: (batch, q-tile)
  dim3 block(256);             // 8 wave32 waves
  din_encoder_kernel<<<grid, block, 0, stream>>>(UE, VM, W1, b1, pa, W2, b2, out);
}
